// PerformerAttention_1219770712660
// MI455X (gfx1250) — matrix operand order repaired and checked by reading, not compiled, not run on hardware
//
#include <hip/hip_runtime.h>

#define HIDDEN   1024
#define HEADS    16
#define HEAD_DIM 64
#define RP       256
#define SEQ      4096
#define BATCH    4
#define TILE_N   128
#define KCHUNK   32

#define CTX_ELEMS  ((size_t)BATCH * HEADS * RP * HEAD_DIM)  // 1048576 f32
#define KSUM_ELEMS ((size_t)BATCH * HEADS * RP)             // 16384 f32
#define XH_ELEMS   ((size_t)BATCH * SEQ * HIDDEN)           // 16777216 f16
#define WT_ELEMS   ((size_t)3 * HIDDEN * HIDDEN)            // 3145728 f16
#define PROJ_ELEMS ((size_t)RP * HEAD_DIM)                  // 16384 f16

typedef __attribute__((ext_vector_type(16))) _Float16 v16h;
typedef __attribute__((ext_vector_type(8)))  _Float16 v8h;
typedef __attribute__((ext_vector_type(8)))  float    v8f;

// ---------------------------------------------------------------------------
// CDNA5 async global->LDS copy (16B per lane), tracked by ASYNCcnt.
// Inline asm (portable across ROCm 7.2 / amdgpu-toolchain per bridge doc).
// ---------------------------------------------------------------------------
__device__ __forceinline__ void async_copy16(_Float16* lds, const _Float16* g) {
  unsigned l = (unsigned)(uintptr_t)lds;   // low 32 bits of generic ptr = LDS offset
  unsigned long long ga = (unsigned long long)(uintptr_t)g;
  asm volatile("global_load_async_to_lds_b128 %0, %1, off"
               :: "v"(l), "v"(ga) : "memory");
}
__device__ __forceinline__ void async_wait0() {
  asm volatile("s_wait_asynccnt 0x0" ::: "memory");
}

// ---------------------------------------------------------------------------
// WMMA fragment loaders (wave32), contiguous k order used for both A and B.
// A (16x32 f16, row-major LDS [M][K], row stride ld):
//   lane l: M = r0 + (l&15); halves 0..15 -> K = kbase + 16*(l>>4) + 0..15
// B (32x16 f16, LDS stored transposed [N][K], row stride ld):
//   lane l: N = n0 + (l&15); halves 0..15 -> K = kbase + 16*(l>>4) + 0..15
// ---------------------------------------------------------------------------
__device__ __forceinline__ v16h ldsFragA(const _Float16* base, int ld, int r0,
                                         int kbase, int lane) {
  int row = r0 + (lane & 15);
  int hi8 = (lane >> 4) * 16;
  const _Float16* p = base + row * ld + kbase + hi8;
  union { v16h v; v8h h[2]; } u;
  u.h[0] = *reinterpret_cast<const v8h*>(p);
  u.h[1] = *reinterpret_cast<const v8h*>(p + 8);
  return u.v;
}

__device__ __forceinline__ v16h ldsFragB(const _Float16* baseT, int ld, int n0,
                                         int kbase, int lane) {
  int n    = n0 + (lane & 15);
  int hi16 = (lane >> 4) * 16;
  return *reinterpret_cast<const v16h*>(baseT + n * ld + kbase + hi16);
}

__device__ __forceinline__ v8f wmma_f16(v16h a, v16h b, v8f c) {
  return __builtin_amdgcn_wmma_f32_16x16x32_f16(false, a, false, b,
                                                (short)0, c, false, false);
}

// ---------------------------------------------------------------------------
// One-shot converts (fast path): f32 -> f16, vectorized 8 elems/thread.
// ---------------------------------------------------------------------------
__global__ void convert_f16_vec(const float* __restrict__ src,
                                _Float16* __restrict__ dst, int n8) {
  int i = blockIdx.x * 256 + threadIdx.x;
  if (i < n8) {
    const float4* s = reinterpret_cast<const float4*>(src) + (size_t)i * 2;
    float4 a = s[0], b = s[1];
    v8h o;
    o[0] = (_Float16)a.x; o[1] = (_Float16)a.y;
    o[2] = (_Float16)a.z; o[3] = (_Float16)a.w;
    o[4] = (_Float16)b.x; o[5] = (_Float16)b.y;
    o[6] = (_Float16)b.z; o[7] = (_Float16)b.w;
    reinterpret_cast<v8h*>(dst)[i] = o;
  }
}

// W [k][c] f32  ->  WhT [c][k] f16 (transposed so tile rows are contiguous)
__global__ void convert_wT(const float* __restrict__ w,
                           _Float16* __restrict__ wT) {
  int i = blockIdx.x * 256 + threadIdx.x;
  if (i < (int)WT_ELEMS) {
    int k = i / (3 * HIDDEN);
    int c = i % (3 * HIDDEN);
    wT[(size_t)c * HIDDEN + k] = (_Float16)w[i];
  }
}

// ---------------------------------------------------------------------------
// Pass A: per (b,h) and 128-row tile:
//   kv = X @ Wkv + bias              (WMMA, K=1024)
//   kp = relu(scale*k @ projT) + eps (WMMA, K=64)
//   ksum += col-sums(kp); ctx += kp^T @ v (WMMA, K=128) via f32 atomics
// F16=1: stage from pre-converted f16 copies with async global->LDS copies.
// ---------------------------------------------------------------------------
template <int F16>
__global__ __launch_bounds__(256, 1)
void performer_pass_a(const float* __restrict__ X, const float* __restrict__ W,
                      const float* __restrict__ Bias,
                      const float* __restrict__ Proj,
                      const _Float16* __restrict__ Xh,
                      const _Float16* __restrict__ WhT,
                      const _Float16* __restrict__ Projh,
                      float* __restrict__ ctx, float* __restrict__ ksum) {
  __shared__ __align__(16) _Float16 projT[RP * HEAD_DIM];   // [m][d]   32 KB
  __shared__ __align__(16) _Float16 xs[TILE_N * KCHUNK];    // [n][k]    8 KB
  __shared__ __align__(16) _Float16 wsT[128 * KCHUNK];      // [c][k]    8 KB
  __shared__ __align__(16) _Float16 ks[TILE_N * HEAD_DIM];  // [n][d]   16 KB
  __shared__ __align__(16) _Float16 vT[HEAD_DIM * TILE_N];  // [d][n]   16 KB
  __shared__ __align__(16) _Float16 kpT[RP * TILE_N];       // [m][n]   64 KB

  const int tid  = threadIdx.x;
  const int lane = tid & 31;
  const int w    = tid >> 5;            // 8 waves
  const int bh   = blockIdx.y;
  const int b    = bh >> 4, h = bh & 15;
  const int n0   = blockIdx.x * TILE_N;
  const int cl   = lane & 15;
  const int hi8  = (lane >> 4) * 8;
  const float scale = 0.125f;           // HEAD_DIM ** -0.5

  const size_t xbase = (size_t)b * SEQ * HIDDEN + (size_t)n0 * HIDDEN;
  const int wk = HIDDEN + h * HEAD_DIM;      // k column base in w_qkv
  const int wv = 2 * HIDDEN + h * HEAD_DIM;  // v column base

  if constexpr (F16) {
    // proj: 32 KB, 2048 x 16B async copies (waited at first chunk)
#pragma unroll
    for (int it = 0; it < 8; ++it) {
      int e = tid + it * 256;
      async_copy16(projT + e * 8, Projh + e * 8);
    }
  } else {
#pragma unroll 4
    for (int it = 0; it < 64; ++it) {
      int e = tid + it * 256;
      projT[e] = (_Float16)Proj[e];
    }
  }

  // ---- GEMM1: kv tile [128 x 128], wave w owns rows 16w..16w+15 -----------
  v8f zero = {};
  v8f acc[8];
#pragma unroll
  for (int i = 0; i < 8; i++) acc[i] = zero;

  for (int kb = 0; kb < HIDDEN; kb += KCHUNK) {
    if constexpr (F16) {
#pragma unroll
      for (int it = 0; it < 2; ++it) {         // X: 512 x 16B
        int e = tid + it * 256;
        int r = e >> 2, seg = (e & 3) * 8;
        async_copy16(xs + r * KCHUNK + seg,
                     Xh + xbase + (size_t)r * HIDDEN + kb + seg);
      }
#pragma unroll
      for (int it = 0; it < 2; ++it) {         // W: 512 x 16B
        int e = tid + it * 256;
        int c = e >> 2, seg = (e & 3) * 8;
        int colg = (c < 64) ? (wk + c) : (wv + (c - 64));
        async_copy16(wsT + c * KCHUNK + seg,
                     WhT + (size_t)colg * HIDDEN + kb + seg);
      }
      async_wait0();
    } else {
#pragma unroll 4
      for (int it = 0; it < 16; ++it) {
        int e = tid + it * 256;
        int r = e >> 5, c = e & 31;
        xs[e] = (_Float16)X[xbase + (size_t)r * HIDDEN + kb + c];
      }
#pragma unroll 4
      for (int it = 0; it < 16; ++it) {
        int e = tid + it * 256;
        int kk = e >> 7, c = e & 127;
        int colg = (c < 64) ? (wk + c) : (wv + (c - 64));
        wsT[c * KCHUNK + kk] = (_Float16)W[(size_t)(kb + kk) * 3 * HIDDEN + colg];
      }
    }
    __syncthreads();
    v16h a = ldsFragA(xs, KCHUNK, w * 16, 0, lane);
#pragma unroll
    for (int ct = 0; ct < 8; ct++) {
      v16h bf = ldsFragB(wsT, KCHUNK, ct * 16, 0, lane);
      acc[ct] = wmma_f16(a, bf, acc[ct]);
    }
    __syncthreads();
  }

  // Epilogue: bias, split k (scaled, row-major) and v (transposed)
#pragma unroll
  for (int ct = 0; ct < 8; ct++) {
    int c = ct * 16 + cl;
    float bias = Bias[(c < 64) ? (wk + c) : (wv + (c - 64))];
    if (ct < 4) {
#pragma unroll
      for (int i = 0; i < 8; i++) {
        int n = w * 16 + i + hi8;
        ks[n * HEAD_DIM + c] = (_Float16)((acc[ct][i] + bias) * scale);
      }
    } else {
      int d = c - 64;
#pragma unroll
      for (int i = 0; i < 8; i++) {
        int n = w * 16 + i + hi8;
        vT[d * TILE_N + n] = (_Float16)(acc[ct][i] + bias);
      }
    }
  }
  __syncthreads();

  // ---- GEMM2: kp = relu(ks @ projT) + eps, store transposed [m][n] --------
  {
    v16h a0 = ldsFragA(ks, HEAD_DIM, w * 16, 0, lane);
    v16h a1 = ldsFragA(ks, HEAD_DIM, w * 16, 32, lane);
#pragma unroll
    for (int mt = 0; mt < 16; mt++) {
      v16h b0 = ldsFragB(projT, HEAD_DIM, mt * 16, 0, lane);
      v16h b1 = ldsFragB(projT, HEAD_DIM, mt * 16, 32, lane);
      v8f c = wmma_f16(a0, b0, zero);
      c = wmma_f16(a1, b1, c);
      int m = mt * 16 + cl;
#pragma unroll
      for (int i = 0; i < 8; i++) {
        int n = w * 16 + i + hi8;
        kpT[m * TILE_N + n] = (_Float16)(fmaxf(c[i], 0.0f) + 1e-3f);
      }
    }
  }
  __syncthreads();

  // ---- k_sum: one column per thread ---------------------------------------
  {
    float s = 0.0f;
#pragma unroll 8
    for (int n = 0; n < TILE_N; n++) s += (float)kpT[tid * TILE_N + n];
    unsafeAtomicAdd(&ksum[bh * RP + tid], s);
  }

  // ---- GEMM3: ctx[m][d] += kp^T @ v; wave w owns m rows 32w..32w+31 -------
  v8f acc3[2][4];
#pragma unroll
  for (int rt = 0; rt < 2; rt++)
#pragma unroll
    for (int dt = 0; dt < 4; dt++) acc3[rt][dt] = zero;

  for (int kc = 0; kc < TILE_N; kc += 32) {
    v16h a0 = ldsFragA(kpT, TILE_N, w * 32, kc, lane);
    v16h a1 = ldsFragA(kpT, TILE_N, w * 32 + 16, kc, lane);
#pragma unroll
    for (int dt = 0; dt < 4; dt++) {
      v16h bf = ldsFragB(vT, TILE_N, dt * 16, kc, lane);
      acc3[0][dt] = wmma_f16(a0, bf, acc3[0][dt]);
      acc3[1][dt] = wmma_f16(a1, bf, acc3[1][dt]);
    }
  }
  float* ctxh = ctx + (size_t)bh * RP * HEAD_DIM;
#pragma unroll
  for (int rt = 0; rt < 2; rt++)
#pragma unroll
    for (int dt = 0; dt < 4; dt++)
#pragma unroll
      for (int i = 0; i < 8; i++) {
        int m = w * 32 + rt * 16 + i + hi8;
        int d = dt * 16 + cl;
        unsafeAtomicAdd(&ctxh[m * HEAD_DIM + d], acc3[rt][dt][i]);
      }
}

// ---------------------------------------------------------------------------
// Pass B: q tile -> q' -> denom -> out = (q' @ ctx) * denom^-1
// ---------------------------------------------------------------------------
template <int F16>
__global__ __launch_bounds__(256, 1)
void performer_pass_b(const float* __restrict__ X, const float* __restrict__ W,
                      const float* __restrict__ Bias,
                      const float* __restrict__ Proj,
                      const _Float16* __restrict__ Xh,
                      const _Float16* __restrict__ WhT,
                      const _Float16* __restrict__ Projh,
                      const float* __restrict__ ctx,
                      const float* __restrict__ ksum,
                      float* __restrict__ out) {
  __shared__ __align__(16) _Float16 projT[RP * HEAD_DIM];   // 32 KB
  __shared__ __align__(16) _Float16 xs[TILE_N * KCHUNK];    //  8 KB
  __shared__ __align__(16) _Float16 wsT[HEAD_DIM * KCHUNK]; //  4 KB
  __shared__ __align__(16) _Float16 qs[TILE_N * HEAD_DIM];  // 16 KB
  __shared__ __align__(16) _Float16 qp[TILE_N * RP];        // 64 KB [n][m]
  __shared__ __align__(16) _Float16 ctxT[HEAD_DIM * RP];    // 32 KB [d][m]
  __shared__ float ksl[RP];
  __shared__ float rden[TILE_N];

  const int tid  = threadIdx.x;
  const int lane = tid & 31;
  const int w    = tid >> 5;
  const int bh   = blockIdx.y;
  const int b    = bh >> 4, h = bh & 15;
  const int n0   = blockIdx.x * TILE_N;
  const int cl   = lane & 15;
  const int hi8  = (lane >> 4) * 8;
  const float scale = 0.125f;

  const size_t xbase = (size_t)b * SEQ * HIDDEN + (size_t)n0 * HIDDEN;
  const int wq = h * HEAD_DIM;  // q column base

  if constexpr (F16) {
#pragma unroll
    for (int it = 0; it < 8; ++it) {
      int e = tid + it * 256;
      async_copy16(projT + e * 8, Projh + e * 8);
    }
  } else {
#pragma unroll 4
    for (int it = 0; it < 64; ++it) {
      int e = tid + it * 256;
      projT[e] = (_Float16)Proj[e];
    }
  }
  const float* ctxh = ctx + (size_t)bh * RP * HEAD_DIM;
#pragma unroll 4
  for (int it = 0; it < 64; ++it) {
    int e = tid + it * 256;
    int m = e >> 6, d = e & 63;
    ctxT[d * RP + m] = (_Float16)ctxh[e];
  }
  if (tid < RP) ksl[tid] = ksum[bh * RP + tid];

  // ---- GEMM1b: q tile [128 x 64] ------------------------------------------
  v8f zero = {};
  v8f acc[4];
#pragma unroll
  for (int i = 0; i < 4; i++) acc[i] = zero;

  for (int kb = 0; kb < HIDDEN; kb += KCHUNK) {
    if constexpr (F16) {
#pragma unroll
      for (int it = 0; it < 2; ++it) {        // X: 512 x 16B
        int e = tid + it * 256;
        int r = e >> 2, seg = (e & 3) * 8;
        async_copy16(xs + r * KCHUNK + seg,
                     Xh + xbase + (size_t)r * HIDDEN + kb + seg);
      }
      {                                        // W: 256 x 16B
        int c = tid >> 2, seg = (tid & 3) * 8;
        async_copy16(wsT + c * KCHUNK + seg,
                     WhT + (size_t)(wq + c) * HIDDEN + kb + seg);
      }
      async_wait0();
    } else {
#pragma unroll 4
      for (int it = 0; it < 16; ++it) {
        int e = tid + it * 256;
        int r = e >> 5, c = e & 31;
        xs[e] = (_Float16)X[xbase + (size_t)r * HIDDEN + kb + c];
      }
#pragma unroll 4
      for (int it = 0; it < 8; ++it) {
        int e = tid + it * 256;
        int kk = e >> 6, c = e & 63;
        wsT[c * KCHUNK + kk] = (_Float16)W[(size_t)(kb + kk) * 3 * HIDDEN + wq + c];
      }
    }
    __syncthreads();
    v16h a = ldsFragA(xs, KCHUNK, w * 16, 0, lane);
#pragma unroll
    for (int ct = 0; ct < 4; ct++) {
      v16h bf = ldsFragB(wsT, KCHUNK, ct * 16, 0, lane);
      acc[ct] = wmma_f16(a, bf, acc[ct]);
    }
    __syncthreads();
  }
#pragma unroll
  for (int ct = 0; ct < 4; ct++) {
    int c = ct * 16 + cl;
    float bias = Bias[wq + c];
#pragma unroll
    for (int i = 0; i < 8; i++) {
      int n = w * 16 + i + hi8;
      qs[n * HEAD_DIM + c] = (_Float16)((acc[ct][i] + bias) * scale);
    }
  }
  __syncthreads();

  // ---- GEMM2b: qp[n][m] = relu(qs @ projT) + eps --------------------------
  {
    v16h a0 = ldsFragA(qs, HEAD_DIM, w * 16, 0, lane);
    v16h a1 = ldsFragA(qs, HEAD_DIM, w * 16, 32, lane);
#pragma unroll
    for (int mt = 0; mt < 16; mt++) {
      v16h b0 = ldsFragB(projT, HEAD_DIM, mt * 16, 0, lane);
      v16h b1 = ldsFragB(projT, HEAD_DIM, mt * 16, 32, lane);
      v8f c = wmma_f16(a0, b0, zero);
      c = wmma_f16(a1, b1, c);
      int m = mt * 16 + cl;
#pragma unroll
      for (int i = 0; i < 8; i++) {
        int n = w * 16 + i + hi8;
        qp[n * RP + m] = (_Float16)(fmaxf(c[i], 0.0f) + 1e-3f);
      }
    }
  }
  __syncthreads();

  // ---- denom: rden[n] = 1 / (qp[n] . ksum) --------------------------------
  if (tid < TILE_N) {
    float s = 0.0f;
#pragma unroll 8
    for (int m = 0; m < RP; m++) s += (float)qp[tid * RP + m] * ksl[m];
    rden[tid] = 1.0f / s;
  }
  __syncthreads();

  // ---- GEMM4: out[n][d] = (qp @ ctx) * rden[n] ----------------------------
  v8f acc4[4];
#pragma unroll
  for (int i = 0; i < 4; i++) acc4[i] = zero;
  for (int kc = 0; kc < RP; kc += 32) {
    v16h a = ldsFragA(qp, RP, w * 16, kc, lane);
#pragma unroll
    for (int dt = 0; dt < 4; dt++) {
      v16h bf = ldsFragB(ctxT, RP, dt * 16, kc, lane);
      acc4[dt] = wmma_f16(a, bf, acc4[dt]);
    }
  }
  float* ob = out + (size_t)b * SEQ * HIDDEN + (size_t)n0 * HIDDEN + h * HEAD_DIM;
#pragma unroll
  for (int dt = 0; dt < 4; dt++) {
    int d = dt * 16 + cl;
#pragma unroll
    for (int i = 0; i < 8; i++) {
      int n = w * 16 + i + hi8;
      ob[(size_t)n * HIDDEN + d] = acc4[dt][i] * rden[n];
    }
  }
}

__global__ void performer_zero_ws(float* __restrict__ p, int n) {
  int i = blockIdx.x * 256 + threadIdx.x;
  if (i < n) p[i] = 0.0f;
}

extern "C" void kernel_launch(void* const* d_in, const int* in_sizes, int n_in,
                              void* d_out, int out_size, void* d_ws,
                              size_t ws_size, hipStream_t stream) {
  const float* X    = (const float*)d_in[0];  // [4,4096,1024]
  const float* W    = (const float*)d_in[1];  // [1024,3072]
  const float* Bq   = (const float*)d_in[2];  // [3072]
  const float* Proj = (const float*)d_in[3];  // [256,64]
  float* out = (float*)d_out;

  float* ctx  = (float*)d_ws;
  float* ksum = ctx + CTX_ELEMS;
  _Float16* Xh    = (_Float16*)(ksum + KSUM_ELEMS);
  _Float16* WhT   = Xh + XH_ELEMS;
  _Float16* Projh = WhT + WT_ELEMS;

  size_t need = (CTX_ELEMS + KSUM_ELEMS) * sizeof(float) +
                (XH_ELEMS + WT_ELEMS + PROJ_ELEMS) * sizeof(_Float16);
  const bool fast = ws_size >= need;

  int ztot = (int)(CTX_ELEMS + KSUM_ELEMS);
  performer_zero_ws<<<(ztot + 255) / 256, 256, 0, stream>>>(ctx, ztot);

  dim3 grid(SEQ / TILE_N, BATCH * HEADS);
  if (fast) {
    int xn8 = (int)(XH_ELEMS / 8);
    convert_f16_vec<<<(xn8 + 255) / 256, 256, 0, stream>>>(X, Xh, xn8);
    int pn8 = (int)(PROJ_ELEMS / 8);
    convert_f16_vec<<<(pn8 + 255) / 256, 256, 0, stream>>>(Proj, Projh, pn8);
    int wn = (int)WT_ELEMS;
    convert_wT<<<(wn + 255) / 256, 256, 0, stream>>>(W, WhT);

    performer_pass_a<1><<<grid, 256, 0, stream>>>(X, W, Bq, Proj, Xh, WhT,
                                                  Projh, ctx, ksum);
    performer_pass_b<1><<<grid, 256, 0, stream>>>(X, W, Bq, Proj, Xh, WhT,
                                                  Projh, ctx, ksum, out);
  } else {
    performer_pass_a<0><<<grid, 256, 0, stream>>>(X, W, Bq, Proj, Xh, WhT,
                                                  Projh, ctx, ksum);
    performer_pass_b<0><<<grid, 256, 0, stream>>>(X, W, Bq, Proj, Xh, WhT,
                                                  Projh, ctx, ksum, out);
  }
}
